// DeepGCN_49289044689219
// MI455X (gfx1250) — matrix operand order repaired and checked by reading, not compiled, not run on hardware
//
#include <hip/hip_runtime.h>

#define DD 128
#define D2 256
#define SLOPE 0.01f
#define EPS_BN 1e-5f
#define EPS_MSG 1e-7f

typedef __attribute__((ext_vector_type(16))) _Float16 v16h;
typedef __attribute__((ext_vector_type(8)))  float    v8f;

// ---------------- small utility kernels ----------------

__global__ void fill_kernel(float* __restrict__ p, float v, int n) {
    int i = blockIdx.x * blockDim.x + threadIdx.x;
    if (i < n) p[i] = v;
}

__device__ inline void atomicMaxFloat(float* addr, float val) {
    // positive floats order as signed ints; negative floats reverse-order as unsigned
    if (val >= 0.f) atomicMax((int*)addr, __float_as_int(val));
    else            atomicMin((unsigned int*)addr, __float_as_uint(val));
}

// column sums for BatchNorm batch statistics; blockDim.x == C (128 or 256)
__global__ void stats_kernel(const float* __restrict__ x, float* __restrict__ sum,
                             float* __restrict__ sq, int N, int C) {
    int c = threadIdx.x;
    float s = 0.f, q = 0.f;
    for (int r = blockIdx.x; r < N; r += gridDim.x) {
        float v = x[(size_t)r * C + c];
        s += v; q += v * v;
    }
    atomicAdd(&sum[c], s);
    atomicAdd(&sq[c],  q);
}

// scale = g*rsqrt(var+eps); shift = b - mu*scale   (one block, blockDim.x == C)
__global__ void finalize_stats(const float* __restrict__ sum, const float* __restrict__ sq,
                               const float* __restrict__ g, const float* __restrict__ b,
                               float* __restrict__ scale, float* __restrict__ shift,
                               int N, int C) {
    int c = threadIdx.x;
    if (c < C) {
        float mu  = sum[c] / (float)N;
        float var = sq[c] / (float)N - mu * mu;
        float rs  = rsqrtf(var + EPS_BN);
        float sc  = g[c] * rs;
        scale[c] = sc;
        shift[c] = b[c] - mu * sc;
    }
}

// h = leaky_relu(bn(x))  elementwise over N*128
__global__ void prenorm_kernel(const float* __restrict__ x, const float* __restrict__ scale,
                               const float* __restrict__ shift, float* __restrict__ h, int nd) {
    int i = blockIdx.x * blockDim.x + threadIdx.x;
    if (i >= nd) return;
    int c = i & (DD - 1);
    float v = x[i] * scale[c] + shift[c];
    h[i] = v > 0.f ? v : SLOPE * v;
}

// ---------------- edge-phase kernels (softmax aggregation) ----------------

__global__ void edge_max_kernel(const float* __restrict__ h, const int* __restrict__ src,
                                const int* __restrict__ dst, const float* __restrict__ ea,
                                const float* __restrict__ We, const float* __restrict__ be,
                                const float* __restrict__ tptr, float* __restrict__ maxbuf,
                                int ED) {
    int idx = blockIdx.x * blockDim.x + threadIdx.x;
    if (idx >= ED) return;
    int e = idx >> 7, c = idx & (DD - 1);
    float t = tptr[0];
    float m = h[(size_t)src[e] * DD + c] + ea[e] * We[c] + be[c];
    m = (m > 0.f ? m : 0.f) + EPS_MSG;
    atomicMaxFloat(&maxbuf[(size_t)dst[e] * DD + c], t * m);
}

// single pass: numer += m*exp(t*m - max), denom += exp(t*m - max)
__global__ void edge_sum_kernel(const float* __restrict__ h, const int* __restrict__ src,
                                const int* __restrict__ dst, const float* __restrict__ ea,
                                const float* __restrict__ We, const float* __restrict__ be,
                                const float* __restrict__ tptr, const float* __restrict__ maxbuf,
                                float* __restrict__ numer, float* __restrict__ denom, int ED) {
    int idx = blockIdx.x * blockDim.x + threadIdx.x;
    if (idx >= ED) return;
    int e = idx >> 7, c = idx & (DD - 1);
    float t = tptr[0];
    float m = h[(size_t)src[e] * DD + c] + ea[e] * We[c] + be[c];
    m = (m > 0.f ? m : 0.f) + EPS_MSG;
    size_t o = (size_t)dst[e] * DD + c;
    float p = __expf(t * m - maxbuf[o]);
    atomicAdd(&numer[o], m * p);
    atomicAdd(&denom[o], p);
}

// agg = numer/denom + h   (nodes with no in-edges: denom==0 -> 0 + h)
__global__ void agg_kernel(const float* __restrict__ numer, const float* __restrict__ denom,
                           const float* __restrict__ h, float* __restrict__ agg, int nd) {
    int i = blockIdx.x * blockDim.x + threadIdx.x;
    if (i >= nd) return;
    float dn = denom[i];
    agg[i] = (dn > 0.f ? numer[i] / dn : 0.f) + h[i];
}

// ---------------- WMMA GEMM (fragment-packed operands) ----------------
// Wave32 fragment maps (ISA 7.12.2):
//   A (16x32 f16): lane half hf, elem j -> K = (j>>3)*16 + 8*hf + (j&7), row = lane&15
//   B (32x16 f16): lane half hf, elem j -> K = (j>>3)*16 + 8*hf + (j&7),               col = lane&15
//   C/D (16x16 f32): elem i -> row = i + 8*hf, col = lane&15

// Pre-pack W (f32 [K, ncols]) into per-fragment-contiguous f16:
//   Wp[(((ct*ktn)+kt)*32 + lane)*16 + j] = W[(kt*32 + (j>>3)*16 + 8*(lane>>4) + (j&7))*ncols + ct*16 + (lane&15)]
__global__ void pack_w_kernel(const float* __restrict__ W, _Float16* __restrict__ Wp,
                              int ktn, int ncols, int total) {
    int idx = blockIdx.x * blockDim.x + threadIdx.x;
    if (idx >= total) return;
    int j    = idx & 15;
    int lane = (idx >> 4) & 31;
    int t    = idx >> 9;
    int kt   = t % ktn;
    int ct   = t / ktn;
    int krow = kt * 32 + ((j >> 3) << 4) + ((lane >> 4) << 3) + (j & 7);
    Wp[idx] = (_Float16)W[(size_t)krow * ncols + ct * 16 + (lane & 15)];
}

// H1 = agg @ W1 + b1 ;  A:[N,128] f32, Bp: packed f16, out:[N,256]
// block = 8 waves -> 16 rows x 128 cols; grid.y selects which 128-col half
__global__ __launch_bounds__(256) void gemm1_wmma(const float* __restrict__ A,
                                                  const _Float16* __restrict__ Bp,
                                                  const float* __restrict__ b1,
                                                  float* __restrict__ H1, int N) {
    __shared__ __align__(32) _Float16 sAh[4][32][16];  // f16 A tile, fragment order (4KB)
    int r0 = blockIdx.x * 16;
    int tid = threadIdx.x;
#pragma unroll
    for (int p = 0; p < 8; ++p) {  // 2048 halves / 256 threads
        int e = p * 256 + tid;
        int j = e & 15, lane = (e >> 4) & 31, kt = e >> 9;
        int hf = lane >> 4, lr = lane & 15;
        int ka = kt * 32 + ((j >> 3) << 4) + (hf << 3) + (j & 7);
        sAh[kt][lane][j] = (_Float16)A[(size_t)(r0 + lr) * DD + ka];
    }
    __syncthreads();
    int wave = tid >> 5, lane = tid & 31;
    int hf = lane >> 4, lr = lane & 15;
    int ct = blockIdx.y * 8 + wave;
    int col = ct * 16 + lr;
    v8f acc;
    float bias = b1[col];
#pragma unroll
    for (int i = 0; i < 8; ++i) acc[i] = bias;
#pragma unroll
    for (int kt = 0; kt < 4; ++kt) {
        v16h a = *(const v16h*)&sAh[kt][lane][0];
        v16h b = *(const v16h*)&Bp[((size_t)(ct * 4 + kt) * 32 + lane) * 16];
        acc = __builtin_amdgcn_wmma_f32_16x16x32_f16(false, a, false, b, (short)0, acc,
                                                     false, false);
    }
#pragma unroll
    for (int i = 0; i < 8; ++i)
        H1[(size_t)(r0 + i + 8 * hf) * D2 + col] = acc[i];
}

// Xout = [residual +] relu(bn(H1)) @ W2 + b2 ; BN+ReLU fused into f16 staging
__global__ __launch_bounds__(256) void gemm2_wmma(const float* __restrict__ H1,
                                                  const float* __restrict__ scale,
                                                  const float* __restrict__ shift,
                                                  const _Float16* __restrict__ Bp,
                                                  const float* __restrict__ b2,
                                                  const float* xres, float* Xout,
                                                  int N, int use_res) {
    __shared__ __align__(32) _Float16 sAh[8][32][16];  // 8KB
    int r0 = blockIdx.x * 16;
    int tid = threadIdx.x;
#pragma unroll
    for (int p = 0; p < 16; ++p) {  // 4096 halves / 256 threads
        int e = p * 256 + tid;
        int j = e & 15, lane = (e >> 4) & 31, kt = e >> 9;
        int hf = lane >> 4, lr = lane & 15;
        int ka = kt * 32 + ((j >> 3) << 4) + (hf << 3) + (j & 7);
        float v = H1[(size_t)(r0 + lr) * D2 + ka];
        v = v * scale[ka] + shift[ka];
        sAh[kt][lane][j] = (_Float16)(v > 0.f ? v : 0.f);  // ReLU fused
    }
    __syncthreads();
    int wave = tid >> 5, lane = tid & 31;
    int hf = lane >> 4, lr = lane & 15;
    int col = wave * 16 + lr;
    v8f acc;
#pragma unroll
    for (int i = 0; i < 8; ++i) {
        int row = r0 + i + 8 * hf;
        acc[i] = b2[col] + (use_res ? xres[(size_t)row * DD + col] : 0.f);
    }
#pragma unroll
    for (int kt = 0; kt < 8; ++kt) {
        v16h a = *(const v16h*)&sAh[kt][lane][0];
        v16h b = *(const v16h*)&Bp[((size_t)(wave * 8 + kt) * 32 + lane) * 16];
        acc = __builtin_amdgcn_wmma_f32_16x16x32_f16(false, a, false, b, (short)0, acc,
                                                     false, false);
    }
#pragma unroll
    for (int i = 0; i < 8; ++i)
        Xout[(size_t)(r0 + i + 8 * hf) * DD + col] = acc[i];
}

// ---------------- pooling ----------------

__global__ void pool_kernel(const float* __restrict__ x, const float* __restrict__ scale,
                            const float* __restrict__ shift, const int* __restrict__ batch,
                            float* __restrict__ pool, int nd) {
    int i = blockIdx.x * blockDim.x + threadIdx.x;
    if (i >= nd) return;
    int r = i >> 7, c = i & (DD - 1);
    float v = x[i] * scale[c] + shift[c];
    v = v > 0.f ? v : SLOPE * v;
    atomicAdd(&pool[(size_t)batch[r] * DD + c], v);
}

__global__ void count_kernel(const int* __restrict__ batch, float* __restrict__ cnt, int N) {
    int r = blockIdx.x * blockDim.x + threadIdx.x;
    if (r < N) atomicAdd(&cnt[batch[r]], 1.f);
}

__global__ void div_kernel(const float* __restrict__ pool, const float* __restrict__ cnt,
                           float* __restrict__ out, int gd) {
    int i = blockIdx.x * blockDim.x + threadIdx.x;
    if (i >= gd) return;
    out[i] = pool[i] / fmaxf(cnt[i >> 7], 1.f);
}

// ---------------- orchestration ----------------

extern "C" void kernel_launch(void* const* d_in, const int* in_sizes, int n_in,
                              void* d_out, int out_size, void* d_ws, size_t ws_size,
                              hipStream_t stream) {
    const float* x_in = (const float*)d_in[0];
    const int*   eidx = (const int*)d_in[1];
    const float* ea   = (const float*)d_in[2];
    const int*   batch= (const int*)d_in[3];
    const float* We   = (const float*)d_in[5];
    const float* be   = (const float*)d_in[6];
    const float* t    = (const float*)d_in[7];
    const float* W1   = (const float*)d_in[8];
    const float* b1   = (const float*)d_in[9];
    const float* g1   = (const float*)d_in[10];
    const float* bt1  = (const float*)d_in[11];
    const float* W2   = (const float*)d_in[12];
    const float* b2   = (const float*)d_in[13];
    const float* bng  = (const float*)d_in[14];
    const float* bnb  = (const float*)d_in[15];

    const int N  = in_sizes[0] / DD;       // 50000
    const int E  = in_sizes[2];            // 800000
    const int L  = in_sizes[7];            // 4
    const int Gn = out_size / DD;          // 64
    const int* src = eidx;
    const int* dst = eidx + E;

    const int ND = N * DD;                 // 6.4M
    const int ED = E * DD;                 // 102.4M
    const int WPACK = DD * D2;             // 32768 f16 elements per weight matrix

    float* ws   = (float*)d_ws;
    float* xbuf = ws;                      // [N,128] layer output
    float* hbuf = xbuf + (size_t)ND;       // [N,128] pre-norm activation
    float* bufA = hbuf + (size_t)ND;       // [N,128] maxbuf, reused as agg
    float* bufB = bufA + (size_t)ND;       // [N,128] numer \ aliased as H1 [N,256]
    float* bufC = bufB + (size_t)ND;       // [N,128] denom /
    float* h1   = bufB;
    float* sum  = bufC + (size_t)ND;       // [256]
    float* sq   = sum + 256;               // [256]
    float* scl  = sq + 256;                // [256]
    float* shf  = scl + 256;               // [256]
    float* pool = shf + 256;               // [G,128]
    float* cnt  = pool + (size_t)Gn * DD;  // [G]
    _Float16* w1p = (_Float16*)(cnt + Gn); // packed f16 W1 (64KB), 32B-aligned offset
    _Float16* w2p = w1p + WPACK;           // packed f16 W2 (64KB)

    const int TB = 256;
    dim3 gND((ND + TB - 1) / TB), gED((ED + TB - 1) / TB);
    dim3 gPK((WPACK + TB - 1) / TB);

    for (int i = 0; i < L; ++i) {
        const float* conv_in;
        if (i == 0) {
            conv_in = x_in;
        } else {
            fill_kernel<<<2, TB, 0, stream>>>(sum, 0.f, 512);  // zero sum+sq
            stats_kernel<<<256, DD, 0, stream>>>(xbuf, sum, sq, N, DD);
            finalize_stats<<<1, DD, 0, stream>>>(sum, sq, bng + i * DD, bnb + i * DD,
                                                 scl, shf, N, DD);
            prenorm_kernel<<<gND, TB, 0, stream>>>(xbuf, scl, shf, hbuf, ND);
            conv_in = hbuf;
        }
        // softmax aggregation accumulators
        fill_kernel<<<gND, TB, 0, stream>>>(bufA, -__builtin_inff(), ND);
        fill_kernel<<<gND, TB, 0, stream>>>(bufB, 0.f, ND);
        fill_kernel<<<gND, TB, 0, stream>>>(bufC, 0.f, ND);
        edge_max_kernel<<<gED, TB, 0, stream>>>(conv_in, src, dst, ea, We + i * DD,
                                                be + i * DD, t + i, bufA, ED);
        edge_sum_kernel<<<gED, TB, 0, stream>>>(conv_in, src, dst, ea, We + i * DD,
                                                be + i * DD, t + i, bufA, bufB, bufC, ED);
        agg_kernel<<<gND, TB, 0, stream>>>(bufB, bufC, conv_in, bufA, ND);
        // MLP: pack weights -> GEMM1 -> BN stats -> GEMM2(+BN+ReLU fused, +residual)
        pack_w_kernel<<<gPK, TB, 0, stream>>>(W1 + (size_t)i * DD * D2, w1p, 4, D2, WPACK);
        pack_w_kernel<<<gPK, TB, 0, stream>>>(W2 + (size_t)i * D2 * DD, w2p, 8, DD, WPACK);
        gemm1_wmma<<<dim3(N / 16, 2), TB, 0, stream>>>(bufA, w1p, b1 + i * D2, h1, N);
        fill_kernel<<<2, TB, 0, stream>>>(sum, 0.f, 512);
        stats_kernel<<<256, D2, 0, stream>>>(h1, sum, sq, N, D2);
        finalize_stats<<<1, D2, 0, stream>>>(sum, sq, g1 + i * D2, bt1 + i * D2,
                                             scl, shf, N, D2);
        gemm2_wmma<<<dim3(N / 16, 1), TB, 0, stream>>>(h1, scl, shf, w2p,
                                                       b2 + i * DD, xbuf, xbuf, N,
                                                       i > 0 ? 1 : 0);
    }

    // final norm (layer-0 BN params) + leaky relu + global mean pool
    fill_kernel<<<2, TB, 0, stream>>>(sum, 0.f, 512);
    stats_kernel<<<256, DD, 0, stream>>>(xbuf, sum, sq, N, DD);
    finalize_stats<<<1, DD, 0, stream>>>(sum, sq, bng, bnb, scl, shf, N, DD);
    int gd = Gn * DD;
    fill_kernel<<<(gd + TB - 1) / TB, TB, 0, stream>>>(pool, 0.f, gd);
    fill_kernel<<<1, TB, 0, stream>>>(cnt, 0.f, Gn);
    pool_kernel<<<gND, TB, 0, stream>>>(xbuf, scl, shf, batch, pool, ND);
    count_kernel<<<(N + TB - 1) / TB, TB, 0, stream>>>(batch, cnt, N);
    div_kernel<<<(gd + TB - 1) / TB, TB, 0, stream>>>(pool, cnt, (float*)d_out, gd);
}
